// RWA_46368466928068
// MI455X (gfx1250) — compile-verified, matrix operand order checked by reading, not run on hardware
//
#include <hip/hip_runtime.h>

// ---------------------------------------------------------------------------
// RWA recurrent net on MI455X (gfx1250), wave32 + WMMA bf16.
// B=128, T=512, F=256, C=1024, K=256.
//
// v3: persistent kernel for all 512 timesteps.
//   - grid = 64 blocks, one per 16-wide n-tile; each block preloads its
//     16x1024 slices of Wg_h / Wa_h into LDS (64 KB) via
//     global_load_async_to_lds_b128 (+ s_wait_asynccnt), reused all steps
//   - x-part weights end up hoisted into (MSB-addressed) VGPRs by the
//     compiler across the t-loop; occupancy is 1 wave/SIMD by design so
//     high VGPR use is free
//   - nin/din/amax (loop-invariant) hoisted into registers before t-loop
//   - atomic sense-reversing grid barrier between steps
//   - h carried in bf16, double-buffered in workspace
// ---------------------------------------------------------------------------

typedef __attribute__((ext_vector_type(16))) __bf16 v16bf;
typedef __attribute__((ext_vector_type(8)))  float  v8f;

#define BB   128
#define TT   512
#define FF   256
#define CC   1024
#define KK   256
#define NBLK 64
#define NTHR 128

__device__ __forceinline__ unsigned short f2bf(float x) {
  unsigned int u = __float_as_uint(x);
  return (unsigned short)((u + 0x7FFFu + ((u >> 16) & 1u)) >> 16);
}

// Load a 16x32 bf16 fragment (A-matrix lane layout; also used for N-major B).
// Lane L: row = L&15 ; lanes 0-15 hold K=0..7 & 16..23, lanes 16-31 hold
// K=8..15 & 24..31 (ISA 7.12.2, 16-bit A 16x32). Works for global or LDS ptr.
__device__ __forceinline__ v16bf load_frag(const unsigned short* base,
                                           int stride, int lane) {
  int row = lane & 15;
  int k0  = (lane >> 4) << 3;  // 0 or 8
  const __bf16* p = reinterpret_cast<const __bf16*>(base + (size_t)row * stride + k0);
  v16bf f;
#pragma unroll
  for (int i = 0; i < 8; ++i) { f[i] = p[i]; f[i + 8] = p[i + 16]; }
  return f;
}

#define WMMA_BF16(A, Bm, Cacc) \
  __builtin_amdgcn_wmma_f32_16x16x32_bf16(false, (A), false, (Bm), (short)0, (Cacc), false, false)

// ---------------------------------------------------------------------------
// Setup kernels
// ---------------------------------------------------------------------------
__global__ void zero_u32_kernel(unsigned* __restrict__ p, int n) {
  int i = blockIdx.x * blockDim.x + threadIdx.x;
  if (i < n) p[i] = 0u;
}

__global__ void convert_bf16_kernel(const float* __restrict__ s,
                                    unsigned short* __restrict__ d, int n) {
  int i = blockIdx.x * blockDim.x + threadIdx.x;
  if (i < n) d[i] = f2bf(s[i]);
}

// dst[n * rows_out + r] = bf16( src[(r + row_off) * cols + n] )  (N-major)
__global__ void transpose_bf16_kernel(const float* __restrict__ s,
                                      unsigned short* __restrict__ d,
                                      int rows_out, int cols, int row_off, int total) {
  int i = blockIdx.x * blockDim.x + threadIdx.x;
  if (i >= total) return;
  int r = i % rows_out;
  int n = i / rows_out;
  d[(size_t)n * rows_out + r] = f2bf(s[(size_t)(r + row_off) * cols + n]);
}

// ---------------------------------------------------------------------------
// Device-wide sense-reversing barrier. bar[0]=arrive count, bar[1]=generation.
// ---------------------------------------------------------------------------
__device__ __forceinline__ void grid_barrier(unsigned* bar) {
  __threadfence();      // make this thread's h writes visible at agent scope
  __syncthreads();
  if (threadIdx.x == 0) {
    unsigned gen  = __hip_atomic_load(&bar[1], __ATOMIC_ACQUIRE, __HIP_MEMORY_SCOPE_AGENT);
    unsigned prev = __hip_atomic_fetch_add(&bar[0], 1u, __ATOMIC_ACQ_REL, __HIP_MEMORY_SCOPE_AGENT);
    if (prev == (unsigned)(NBLK - 1)) {
      __hip_atomic_store(&bar[0], 0u, __ATOMIC_RELAXED, __HIP_MEMORY_SCOPE_AGENT);
      __hip_atomic_store(&bar[1], gen + 1u, __ATOMIC_RELEASE, __HIP_MEMORY_SCOPE_AGENT);
    } else {
      while (__hip_atomic_load(&bar[1], __ATOMIC_ACQUIRE, __HIP_MEMORY_SCOPE_AGENT) == gen)
        __builtin_amdgcn_s_sleep(1);
    }
  }
  __syncthreads();
}

// ---------------------------------------------------------------------------
// Persistent recurrence kernel: 64 blocks x 128 threads (4 waves).
// Block b owns n-tile b (cells n0..n0+15). Wave w computes m-tiles w and w+4.
// ---------------------------------------------------------------------------
__global__ __launch_bounds__(NTHR) void
rwa_persistent_kernel(const unsigned short* __restrict__ xbf,
                      unsigned short* __restrict__ hb0,
                      unsigned short* __restrict__ hb1,
                      const unsigned short* __restrict__ WuT,
                      const unsigned short* __restrict__ WgxT,
                      const unsigned short* __restrict__ WghT,
                      const unsigned short* __restrict__ WaxT,
                      const unsigned short* __restrict__ WahT,
                      const float* __restrict__ bg,
                      const float* __restrict__ bu,
                      const float* __restrict__ nin,
                      const float* __restrict__ din,
                      const float* __restrict__ amax,
                      unsigned* __restrict__ bar) {
  const int lane = threadIdx.x & 31;
  const int wave = threadIdx.x >> 5;          // 0..3
  const int n0   = blockIdx.x << 4;           // n-tile origin (0..1008)
  const int m0A  = wave << 4;                 // batch rows  0..48
  const int m0B  = (wave << 4) + 64;          // batch rows 64..112

  // LDS: this block's 16-column slices of Wg_h and Wa_h (bf16, N-major ->
  // 16 consecutive rows of 1024 are one contiguous 32 KB chunk each).
  __shared__ unsigned short smem[2 * 16 * CC];   // 64 KB
  {
    // Async DMA of both 32 KB slices into LDS: each lane copies 16 B per
    // issued instruction (one wave-wide instr moves 512 B); 16 instrs per
    // matrix per thread-slot => ASYNCcnt stays well under its 6-bit limit.
    unsigned ldsbase = (unsigned)(size_t)(void*)smem;   // LDS aperture: low 32 bits = LDS offset
    const char* gsrcG = (const char*)(WghT + (size_t)n0 * CC);
    const char* gsrcA = (const char*)(WahT + (size_t)n0 * CC);
    const int nchunks = (16 * CC * 2) / 16;             // 2048 x 16B per matrix
    for (int i = threadIdx.x; i < nchunks; i += NTHR) {
      unsigned long long gaG = (unsigned long long)(size_t)(gsrcG + (size_t)i * 16);
      unsigned long long gaA = (unsigned long long)(size_t)(gsrcA + (size_t)i * 16);
      unsigned laG = ldsbase + (unsigned)(i * 16);
      unsigned laA = ldsbase + (unsigned)(16 * CC * 2) + (unsigned)(i * 16);
      asm volatile("global_load_async_to_lds_b128 %0, %1, off"
                   :: "v"(laG), "v"(gaG) : "memory");
      asm volatile("global_load_async_to_lds_b128 %0, %1, off"
                   :: "v"(laA), "v"(gaA) : "memory");
    }
    asm volatile("s_wait_asynccnt 0" ::: "memory");
  }
  __syncthreads();

  const int  mofs = (lane >> 4) << 3;
  const int  n    = n0 + (lane & 15);
  const float bgn = bg[n];
  const float bun = bu[n];

  // Loop-invariant epilogue state: n, d, a_max are never reassigned in the
  // reference loop -> load once, keep in registers (occupancy is 1 wave/SIMD
  // by design, VGPRs are plentiful).
  float niR[16], diR[16], amR[16];
#pragma unroll
  for (int half = 0; half < 2; ++half) {
#pragma unroll
    for (int i = 0; i < 8; ++i) {
      int m = (half ? m0B : m0A) + i + mofs;
      size_t idx = (size_t)m * CC + n;
      niR[half * 8 + i] = nin[idx];
      diR[half * 8 + i] = din[idx];
      amR[half * 8 + i] = amax[idx];
    }
  }

  for (int t = 0; t < TT; ++t) {
    const unsigned short* hc = (t & 1) ? hb1 : hb0;
    unsigned short*       hn = (t & 1) ? hb0 : hb1;

    v8f aG0 = {}; v8f aA0 = {}; v8f aU0 = {};
    v8f aG1 = {}; v8f aA1 = {}; v8f aU1 = {};

    // ---- x-dependent part: K = 0..255 (weights loop-invariant ->
    //      compiler keeps these fragments in MSB-addressed VGPRs) ----
    const unsigned short* xr0 = xbf + (size_t)m0A * (TT * FF) + (size_t)t * FF;
    const unsigned short* xr1 = xbf + (size_t)m0B * (TT * FF) + (size_t)t * FF;
#pragma unroll
    for (int k = 0; k < FF; k += 32) {
      v16bf bG = load_frag(WgxT + (size_t)n0 * FF + k, FF, lane);
      v16bf bA = load_frag(WaxT + (size_t)n0 * FF + k, FF, lane);
      v16bf bU = load_frag(WuT  + (size_t)n0 * FF + k, FF, lane);
      v16bf a0 = load_frag(xr0 + k, TT * FF, lane);
      v16bf a1 = load_frag(xr1 + k, TT * FF, lane);
      aG0 = WMMA_BF16(a0, bG, aG0);
      aG1 = WMMA_BF16(a1, bG, aG1);
      aA0 = WMMA_BF16(a0, bA, aA0);
      aA1 = WMMA_BF16(a1, bA, aA1);
      aU0 = WMMA_BF16(a0, bU, aU0);
      aU1 = WMMA_BF16(a1, bU, aU1);
    }

    // prefetch next timestep's x rows (overlaps with h-part math)
    if (t + 1 < TT) {
      const unsigned short* nx0 =
          xbf + (size_t)(m0A + (lane & 15)) * (TT * FF) + (size_t)(t + 1) * FF;
      __builtin_prefetch(nx0, 0, 0);
      __builtin_prefetch(nx0 + (size_t)64 * (TT * FF), 0, 0);
    }

    // ---- h-dependent part: K = 0..1023 (weights in LDS) ----
    const unsigned short* hr0 = hc + (size_t)m0A * CC;
    const unsigned short* hr1 = hc + (size_t)m0B * CC;
#pragma unroll 4
    for (int k = 0; k < CC; k += 32) {
      v16bf bG = load_frag(smem + k, CC, lane);
      v16bf bA = load_frag(smem + 16 * CC + k, CC, lane);
      v16bf a0 = load_frag(hr0 + k, CC, lane);
      v16bf a1 = load_frag(hr1 + k, CC, lane);
      aG0 = WMMA_BF16(a0, bG, aG0);
      aG1 = WMMA_BF16(a1, bG, aG1);
      aA0 = WMMA_BF16(a0, bA, aA0);
      aA1 = WMMA_BF16(a1, bA, aA1);
    }

    // ---- epilogue: z = u*tanh(g); running-max exp blend; h = tanh(n/d) ----
#pragma unroll
    for (int half = 0; half < 2; ++half) {
      const int m0 = half ? m0B : m0A;
      const v8f& accG = half ? aG1 : aG0;
      const v8f& accA = half ? aA1 : aA0;
      const v8f& accU = half ? aU1 : aU0;
#pragma unroll
      for (int i = 0; i < 8; ++i) {
        int m = m0 + i + mofs;
        size_t idx = (size_t)m * CC + n;
        float g = accG[i] + bgn;
        float u = accU[i] + bun;
        float a = accA[i];
        float z = u * tanhf(g);
        float am    = amR[half * 8 + i];
        float anew  = fmaxf(am, a);
        float e_old = expf(am - anew);
        float e_t   = expf(a - anew);
        float n_t   = niR[half * 8 + i] * e_old + z * e_t;
        float d_t   = diR[half * 8 + i] * e_old + e_t;
        hn[idx] = f2bf(tanhf(n_t / d_t));
      }
    }

    grid_barrier(bar);
  }
}

// ---------------------------------------------------------------------------
// Final GEMM: out[B,K] = h[B,C] @ Wo[C,K] + bo. 128 tiles, 32 blocks x 4 waves.
// ---------------------------------------------------------------------------
__global__ void out_gemm_kernel(const unsigned short* __restrict__ hbf,
                                const unsigned short* __restrict__ WoT,
                                const float* __restrict__ bo,
                                float* __restrict__ out) {
  const int lane = threadIdx.x & 31;
  const int wave = threadIdx.x >> 5;
  const int tile = blockIdx.x * 4 + wave;       // 0..127
  const int m0 = (tile & 7) << 4;
  const int n0 = (tile >> 3) << 4;

  v8f acc = {};
#pragma unroll 4
  for (int k = 0; k < CC; k += 32) {
    v16bf a = load_frag(hbf + (size_t)m0 * CC + k, CC, lane);
    v16bf b = load_frag(WoT + (size_t)n0 * CC + k, CC, lane);
    acc = WMMA_BF16(a, b, acc);
  }

  const int mofs = (lane >> 4) << 3;
  const int n    = n0 + (lane & 15);
#pragma unroll
  for (int i = 0; i < 8; ++i) {
    int m = m0 + i + mofs;
    out[(size_t)m * KK + n] = acc[i] + bo[n];
  }
}

// ---------------------------------------------------------------------------
extern "C" void kernel_launch(void* const* d_in, const int* in_sizes, int n_in,
                              void* d_out, int out_size, void* d_ws, size_t ws_size,
                              hipStream_t stream) {
  (void)in_sizes; (void)n_in; (void)out_size; (void)ws_size;

  const float* x    = (const float*)d_in[0];   // [B,T,F]
  const float* nin  = (const float*)d_in[1];   // [B,C]
  const float* din  = (const float*)d_in[2];   // [B,C]
  const float* h0   = (const float*)d_in[3];   // [B,C]
  const float* amax = (const float*)d_in[4];   // [B,C]
  const float* Wg   = (const float*)d_in[5];   // [F+C, C]
  const float* bg   = (const float*)d_in[6];   // [C]
  const float* Wu   = (const float*)d_in[7];   // [F, C]
  const float* bu   = (const float*)d_in[8];   // [C]
  const float* Wa   = (const float*)d_in[9];   // [F+C, C]
  const float* Wo   = (const float*)d_in[10];  // [C, K]
  const float* bo   = (const float*)d_in[11];  // [K]
  float* out = (float*)d_out;                  // [B, K]

  // Workspace carve, 256B aligned
  char* w = (char*)d_ws;
  auto carve = [&](size_t bytes) -> char* {
    char* p = w;
    w += (bytes + 255) & ~(size_t)255;
    return p;
  };
  unsigned short* xbf  = (unsigned short*)carve((size_t)BB * TT * FF * 2);  // 33.5 MB
  unsigned short* hA   = (unsigned short*)carve((size_t)BB * CC * 2);
  unsigned short* hBuf = (unsigned short*)carve((size_t)BB * CC * 2);
  unsigned short* WuT  = (unsigned short*)carve((size_t)CC * FF * 2);  // [C][F]
  unsigned short* WgxT = (unsigned short*)carve((size_t)CC * FF * 2);  // [C][F]
  unsigned short* WghT = (unsigned short*)carve((size_t)CC * CC * 2);  // [C][C]
  unsigned short* WaxT = (unsigned short*)carve((size_t)CC * FF * 2);
  unsigned short* WahT = (unsigned short*)carve((size_t)CC * CC * 2);
  unsigned short* WoT  = (unsigned short*)carve((size_t)KK * CC * 2);  // [K][C]
  unsigned*       bar  = (unsigned*)carve(256);

  // --- init barrier + one-time conversions ---
  zero_u32_kernel<<<1, 64, 0, stream>>>(bar, 64);
  {
    int ncv = BB * TT * FF;
    convert_bf16_kernel<<<(ncv + 255) / 256, 256, 0, stream>>>(x, xbf, ncv);
  }
  {
    int ncv = BB * CC;
    convert_bf16_kernel<<<(ncv + 255) / 256, 256, 0, stream>>>(h0, hA, ncv);
  }
  {
    int tot = FF * CC;  // rows_out=F, cols=C
    transpose_bf16_kernel<<<(tot + 255) / 256, 256, 0, stream>>>(Wu, WuT, FF, CC, 0, tot);
    transpose_bf16_kernel<<<(tot + 255) / 256, 256, 0, stream>>>(Wg, WgxT, FF, CC, 0, tot);
    transpose_bf16_kernel<<<(tot + 255) / 256, 256, 0, stream>>>(Wa, WaxT, FF, CC, 0, tot);
  }
  {
    int tot = CC * CC;  // rows_out=C, cols=C, row_off=F
    transpose_bf16_kernel<<<(tot + 255) / 256, 256, 0, stream>>>(Wg, WghT, CC, CC, FF, tot);
    transpose_bf16_kernel<<<(tot + 255) / 256, 256, 0, stream>>>(Wa, WahT, CC, CC, FF, tot);
  }
  {
    int tot = CC * KK;  // rows_out=C, cols=K
    transpose_bf16_kernel<<<(tot + 255) / 256, 256, 0, stream>>>(Wo, WoT, CC, KK, 0, tot);
  }

  // --- all 512 recurrence steps in one persistent kernel ---
  // t even: read hA  -> write hBuf ; t odd: read hBuf -> write hA.
  // Last step (t=511, odd) writes hA => final h lives in hA.
  rwa_persistent_kernel<<<NBLK, NTHR, 0, stream>>>(xbf, hA, hBuf,
                                                   WuT, WgxT, WghT, WaxT, WahT,
                                                   bg, bu, nin, din, amax, bar);

  // --- output projection ---
  out_gemm_kernel<<<32, 128, 0, stream>>>(hA, WoT, bo, out);
}
